// MessagePassingLayer_27333171872051
// MI455X (gfx1250) — matrix operand order repaired and checked by reading, not compiled, not run on hardware
//
#include <hip/hip_runtime.h>
#include <hip/hip_bf16.h>
#include <stdint.h>

// ---------------------------------------------------------------------------
// MessagePassingLayer for MI455X (gfx1250), bf16 WMMA path.
// B=2, N=20000, H=256, E=160000, M=512, 2H=512.
// Edge kernel: 64-edge tiles (4x B-fragment reuse), node kernel: 32-node tiles.
// ---------------------------------------------------------------------------

#define GB 2
#define GN 20000
#define GH 256
#define GE 160000
#define GM 512
#define K2H 512   // 2*H

typedef __bf16 bf16_t;
typedef __attribute__((ext_vector_type(16))) __bf16 v16bf;
typedef __attribute__((ext_vector_type(8)))  __bf16 v8bf;
typedef __attribute__((ext_vector_type(8)))  float  v8f;

__device__ __forceinline__ bf16_t f2bf(float f) {
    // round-to-nearest-even float32 -> bf16
    uint32_t u = __builtin_bit_cast(uint32_t, f);
    uint32_t r = (u + 0x7FFFu + ((u >> 16) & 1u)) >> 16;
    uint16_t s = (uint16_t)r;
    return __builtin_bit_cast(bf16_t, s);
}

// Assemble a 16-element bf16 A-fragment from two 8-element (16B) chunks.
// ISA 16-bit A-matrix layout: per lane, two groups of 8 consecutive K.
__device__ __forceinline__ v16bf frag_a(const bf16_t* lo, const bf16_t* hi) {
    v8bf a = *reinterpret_cast<const v8bf*>(lo);
    v8bf b = *reinterpret_cast<const v8bf*>(hi);
    v16bf r;
#pragma unroll
    for (int i = 0; i < 8; ++i) { r[i] = a[i]; r[i + 8] = b[i]; }
    return r;
}

// ---------------------------------------------------------------------------
// Preprocessing kernels
// ---------------------------------------------------------------------------
__global__ void cvt_f32_bf16_kernel(const float* __restrict__ in,
                                    bf16_t* __restrict__ out, int n) {
    int i = blockIdx.x * blockDim.x + threadIdx.x;
    if (i < n) out[i] = f2bf(in[i]);
}

// in: [K][Nc] f32 row-major  ->  out: [Nc][K] bf16 row-major (transposed)
__global__ void transpose_w_kernel(const float* __restrict__ in,
                                   bf16_t* __restrict__ out, int K, int Nc) {
    int i = blockIdx.x * blockDim.x + threadIdx.x;
    if (i < K * Nc) {
        int k = i / Nc;
        int n = i - k * Nc;
        out[(size_t)n * K + k] = f2bf(in[i]);
    }
}

__global__ void zero_f32_kernel(float* __restrict__ p, int n) {
    int i = blockIdx.x * blockDim.x + threadIdx.x;
    if (i < n) p[i] = 0.0f;
}

// ---------------------------------------------------------------------------
// Edge (message) kernel: 64 edges per block, 256 threads (8 waves).
//   x = [h[src] | h[dst]] (64 x 512 bf16) -> relu(x@W1+b1) -> @W2+b2
//   -> atomic scatter-add into agg[b][dst][:]
// Each wave owns 4 column-tiles (stage B) / 2 column-tiles (stage C) and
// keeps 4 row-tile accumulators, reusing every global B fragment 4x.
// ---------------------------------------------------------------------------
__global__ void __launch_bounds__(256)
mpl_edge_kernel(const bf16_t* __restrict__ hb,
                const long long* __restrict__ eidx,
                const bf16_t* __restrict__ w1T,   // [M][2H]
                const float*  __restrict__ b1,
                const bf16_t* __restrict__ w2T,   // [H][M]
                const float*  __restrict__ b2,
                float* __restrict__ agg) {
    extern __shared__ char smem[];
    bf16_t* xs  = (bf16_t*)smem;                    // 64 x 512 bf16 = 64 KB
    bf16_t* hsd = (bf16_t*)(smem + 64 * K2H * 2);   // 64 x 512 bf16 = 64 KB
    int* ssrc   = (int*)(smem + 2 * 64 * K2H * 2);
    int* sdst   = ssrc + 64;

    const int tid  = threadIdx.x;
    const int b    = blockIdx.y;
    const int e0   = blockIdx.x * 64;
    const int lane = tid & 31;
    const int wave = tid >> 5;
    const int cwl  = lane & 15;   // column / row-within-tile for this lane
    const int kh   = lane >> 4;   // K-half select

    if (tid < 64) {
        ssrc[tid] = (int)eidx[e0 + tid];
        sdst[tid] = (int)eidx[GE + e0 + tid];
    }
    __syncthreads();

    // -------- Stage A: gather x tile (64 rows x 512 cols) in 8-bf16 chunks
    for (int c = tid; c < 64 * 64; c += 256) {
        int row  = c >> 6;
        int f    = (c & 63) * 8;
        int node = (f < GH) ? ssrc[row] : sdst[row];
        int hf   = f & (GH - 1);
        *reinterpret_cast<v8bf*>(&xs[row * K2H + f]) =
            *reinterpret_cast<const v8bf*>(hb + ((size_t)b * GN + node) * GH + hf);
    }
    __syncthreads();

    const v8f vzero = {};

    // -------- Stage B: hidden = relu(x @ W1 + b1); 32 col-tiles / 8 waves
#pragma unroll
    for (int j = 0; j < 4; ++j) {
        const int cb = (wave * 4 + j) * 16;
        const bf16_t* wcol = w1T + (size_t)(cb + cwl) * K2H;
        __builtin_prefetch(wcol, 0, 0);
        v8f acc[4] = {vzero, vzero, vzero, vzero};
#pragma unroll 4
        for (int ks = 0; ks < K2H / 32; ++ks) {
            const int K0 = ks * 32;
            v16bf w = *reinterpret_cast<const v16bf*>(wcol + K0 + kh * 16);
#pragma unroll
            for (int r = 0; r < 4; ++r) {
                const int arow = r * 16 + cwl;
                v16bf a = frag_a(&xs[arow * K2H + K0 + kh * 16],
                                 &xs[arow * K2H + K0 + kh * 16 + 8]);
                acc[r] = __builtin_amdgcn_wmma_f32_16x16x32_bf16(
                    false, a, false, w, (short)0, acc[r], false, false);
            }
        }
        const float bias = b1[cb + cwl];
#pragma unroll
        for (int r = 0; r < 4; ++r)
#pragma unroll
            for (int v = 0; v < 8; ++v) {
                float t = acc[r][v] + bias;
                t = t > 0.0f ? t : 0.0f;
                hsd[(r * 16 + kh * 8 + v) * GM + cb + cwl] = f2bf(t);
            }
    }
    __syncthreads();

    // -------- Stage C: msg = hidden @ W2 + b2, scatter-add into agg[dst]
#pragma unroll
    for (int j = 0; j < 2; ++j) {
        const int cb = (wave * 2 + j) * 16;
        const bf16_t* wcol = w2T + (size_t)(cb + cwl) * GM;
        __builtin_prefetch(wcol, 0, 0);
        v8f acc[4] = {vzero, vzero, vzero, vzero};
#pragma unroll 4
        for (int ks = 0; ks < GM / 32; ++ks) {
            const int K0 = ks * 32;
            v16bf w = *reinterpret_cast<const v16bf*>(wcol + K0 + kh * 16);
#pragma unroll
            for (int r = 0; r < 4; ++r) {
                const int arow = r * 16 + cwl;
                v16bf a = frag_a(&hsd[arow * GM + K0 + kh * 16],
                                 &hsd[arow * GM + K0 + kh * 16 + 8]);
                acc[r] = __builtin_amdgcn_wmma_f32_16x16x32_bf16(
                    false, a, false, w, (short)0, acc[r], false, false);
            }
        }
        const float bias = b2[cb + cwl];
#pragma unroll
        for (int r = 0; r < 4; ++r)
#pragma unroll
            for (int v = 0; v < 8; ++v) {
                const int m = r * 16 + kh * 8 + v;
                const int d = sdst[m];
                atomicAdd(agg + ((size_t)b * GN + d) * GH + cb + cwl,
                          acc[r][v] + bias);
            }
    }
}

// ---------------------------------------------------------------------------
// Node (update) kernel: 32 nodes per block.
//   x = [h | agg] -> relu(x@U1+c1) -> @U2+c2 ; out = h + upd
// ---------------------------------------------------------------------------
__global__ void __launch_bounds__(256)
mpl_node_kernel(const float*  __restrict__ hin,
                const bf16_t* __restrict__ hb,
                const float*  __restrict__ agg,
                const bf16_t* __restrict__ w1T,   // [M][2H]
                const float*  __restrict__ b1,
                const bf16_t* __restrict__ w2T,   // [H][M]
                const float*  __restrict__ b2,
                float* __restrict__ out) {
    extern __shared__ char smem[];
    bf16_t* xs  = (bf16_t*)smem;                    // 32 x 512 bf16 = 32 KB
    bf16_t* hsd = (bf16_t*)(smem + 32 * K2H * 2);   // 32 x 512 bf16 = 32 KB

    const int tid  = threadIdx.x;
    const int b    = blockIdx.y;
    const int n0   = blockIdx.x * 32;
    const int lane = tid & 31;
    const int wave = tid >> 5;
    const int cwl  = lane & 15;
    const int kh   = lane >> 4;

    // -------- Stage A: build [h | agg] tile (32 rows x 512 cols)
    for (int c = tid; c < 32 * 64; c += 256) {
        int row  = c >> 6;
        int f    = (c & 63) * 8;
        int node = n0 + row;
        if (f < GH) {
            *reinterpret_cast<v8bf*>(&xs[row * K2H + f]) =
                *reinterpret_cast<const v8bf*>(hb + ((size_t)b * GN + node) * GH + f);
        } else {
            const float* ap = agg + ((size_t)b * GN + node) * GH + (f - GH);
#pragma unroll
            for (int i = 0; i < 8; ++i)
                xs[row * K2H + f + i] = f2bf(ap[i]);
        }
    }
    __syncthreads();

    const v8f vzero = {};

    // -------- Stage B: hidden
#pragma unroll
    for (int j = 0; j < 4; ++j) {
        const int cb = (wave * 4 + j) * 16;
        const bf16_t* wcol = w1T + (size_t)(cb + cwl) * K2H;
        __builtin_prefetch(wcol, 0, 0);
        v8f acc[2] = {vzero, vzero};
#pragma unroll 4
        for (int ks = 0; ks < K2H / 32; ++ks) {
            const int K0 = ks * 32;
            v16bf w = *reinterpret_cast<const v16bf*>(wcol + K0 + kh * 16);
#pragma unroll
            for (int r = 0; r < 2; ++r) {
                const int arow = r * 16 + cwl;
                v16bf a = frag_a(&xs[arow * K2H + K0 + kh * 16],
                                 &xs[arow * K2H + K0 + kh * 16 + 8]);
                acc[r] = __builtin_amdgcn_wmma_f32_16x16x32_bf16(
                    false, a, false, w, (short)0, acc[r], false, false);
            }
        }
        const float bias = b1[cb + cwl];
#pragma unroll
        for (int r = 0; r < 2; ++r)
#pragma unroll
            for (int v = 0; v < 8; ++v) {
                float t = acc[r][v] + bias;
                t = t > 0.0f ? t : 0.0f;
                hsd[(r * 16 + kh * 8 + v) * GM + cb + cwl] = f2bf(t);
            }
    }
    __syncthreads();

    // -------- Stage C: out = h + hidden @ U2 + c2
#pragma unroll
    for (int j = 0; j < 2; ++j) {
        const int cb = (wave * 2 + j) * 16;
        const bf16_t* wcol = w2T + (size_t)(cb + cwl) * GM;
        __builtin_prefetch(wcol, 0, 0);
        v8f acc[2] = {vzero, vzero};
#pragma unroll 4
        for (int ks = 0; ks < GM / 32; ++ks) {
            const int K0 = ks * 32;
            v16bf w = *reinterpret_cast<const v16bf*>(wcol + K0 + kh * 16);
#pragma unroll
            for (int r = 0; r < 2; ++r) {
                const int arow = r * 16 + cwl;
                v16bf a = frag_a(&hsd[arow * GM + K0 + kh * 16],
                                 &hsd[arow * GM + K0 + kh * 16 + 8]);
                acc[r] = __builtin_amdgcn_wmma_f32_16x16x32_bf16(
                    false, a, false, w, (short)0, acc[r], false, false);
            }
        }
        const float bias = b2[cb + cwl];
#pragma unroll
        for (int r = 0; r < 2; ++r)
#pragma unroll
            for (int v = 0; v < 8; ++v) {
                const int m = r * 16 + kh * 8 + v;
                const size_t o = ((size_t)b * GN + (n0 + m)) * GH + cb + cwl;
                out[o] = hin[o] + acc[r][v] + bias;
            }
    }
}

// ---------------------------------------------------------------------------
// Launch
// ---------------------------------------------------------------------------
extern "C" void kernel_launch(void* const* d_in, const int* in_sizes, int n_in,
                              void* d_out, int out_size, void* d_ws, size_t ws_size,
                              hipStream_t stream) {
    const float*     h      = (const float*)d_in[0];
    const long long* eidx   = (const long long*)d_in[1];
    const float*     msg_w1 = (const float*)d_in[2];
    const float*     msg_b1 = (const float*)d_in[3];
    const float*     msg_w2 = (const float*)d_in[4];
    const float*     msg_b2 = (const float*)d_in[5];
    const float*     upd_w1 = (const float*)d_in[6];
    const float*     upd_b1 = (const float*)d_in[7];
    const float*     upd_w2 = (const float*)d_in[8];
    const float*     upd_b2 = (const float*)d_in[9];
    float*           out    = (float*)d_out;

    // workspace layout
    char* ws = (char*)d_ws;
    size_t off = 0;
    bf16_t* hb    = (bf16_t*)(ws + off); off += (size_t)GB * GN * GH * 2;      // 20.48 MB
    float*  agg   = (float*) (ws + off); off += (size_t)GB * GN * GH * 4;      // 40.96 MB
    bf16_t* mw1T  = (bf16_t*)(ws + off); off += (size_t)GM * K2H * 2;          // 512 KB
    bf16_t* mw2T  = (bf16_t*)(ws + off); off += (size_t)GH * GM * 2;           // 256 KB
    bf16_t* uw1T  = (bf16_t*)(ws + off); off += (size_t)GM * K2H * 2;
    bf16_t* uw2T  = (bf16_t*)(ws + off); off += (size_t)GH * GM * 2;

    const int nh = GB * GN * GH;   // 10,240,000

    cvt_f32_bf16_kernel<<<(nh + 255) / 256, 256, 0, stream>>>(h, hb, nh);
    transpose_w_kernel<<<(K2H * GM + 255) / 256, 256, 0, stream>>>(msg_w1, mw1T, K2H, GM);
    transpose_w_kernel<<<(GM * GH + 255) / 256, 256, 0, stream>>>(msg_w2, mw2T, GM, GH);
    transpose_w_kernel<<<(K2H * GM + 255) / 256, 256, 0, stream>>>(upd_w1, uw1T, K2H, GM);
    transpose_w_kernel<<<(GM * GH + 255) / 256, 256, 0, stream>>>(upd_w2, uw2T, GM, GH);
    zero_f32_kernel<<<(nh + 255) / 256, 256, 0, stream>>>(agg, nh);

    // edge kernel: 64 edges/block, LDS = 2*64KB + 2*64*4B index arrays
    const size_t edge_lds = (size_t)2 * 64 * K2H * 2 + 2 * 64 * sizeof(int);
    dim3 eg(GE / 64, GB);
    mpl_edge_kernel<<<eg, 256, edge_lds, stream>>>(hb, eidx, mw1T, msg_b1,
                                                   mw2T, msg_b2, agg);

    // node kernel: 32 nodes/block, LDS = 2*32KB
    const size_t node_lds = (size_t)2 * 32 * K2H * 2;
    dim3 ng(GN / 32, GB);
    mpl_node_kernel<<<ng, 256, node_lds, stream>>>(h, hb, agg, uw1T, upd_b1,
                                                   uw2T, upd_b2, out);
}
